// AttentionBlock_15556371546709
// MI455X (gfx1250) — matrix operand order repaired and checked by reading, not compiled, not run on hardware
//
#include <hip/hip_runtime.h>

typedef __attribute__((ext_vector_type(16))) _Float16 v16h;
typedef __attribute__((ext_vector_type(8)))  _Float16 v8h;
typedef __attribute__((ext_vector_type(8)))  float    v8f;
typedef __attribute__((ext_vector_type(4)))  float    v4f;

#define BATCH 4
#define CH    256
#define HW    4096          // 64*64 tokens
#define NGRP  16
#define CPG   16            // channels per group
#define OC3   768           // 3*CH
#define QW    8             // waves per flash block

union H16 { v16h v; v8h h2[2]; _Float16 e[16]; };

__device__ __forceinline__ v8f wmma16(v16h a, v16h b, v8f c) {
    // D = A(16x32 f16) * B(32x16 f16) + C(16x16 f32)
    return __builtin_amdgcn_wmma_f32_16x16x32_f16(false, a, false, b,
                                                  (short)0, c, false, false);
}

// async global->LDS DMA: 16B per lane, same inst offset applies to both sides
__device__ __forceinline__ void async_b128(unsigned lds_off, unsigned long long gaddr) {
    asm volatile("global_load_async_to_lds_b128 %0, %1, off offset:0"
                 :: "v"(lds_off), "v"(gaddr) : "memory");
    asm volatile("global_load_async_to_lds_b128 %0, %1, off offset:16"
                 :: "v"(lds_off), "v"(gaddr) : "memory");
    asm volatile("global_load_async_to_lds_b128 %0, %1, off offset:32"
                 :: "v"(lds_off), "v"(gaddr) : "memory");
    asm volatile("global_load_async_to_lds_b128 %0, %1, off offset:48"
                 :: "v"(lds_off), "v"(gaddr) : "memory");
}

// ---------------------------------------------------------------- GN stats
__global__ __launch_bounds__(256) void gn_stats_kernel(
        const float* __restrict__ x, float* __restrict__ mu,
        float* __restrict__ rstd) {
    int bg = blockIdx.x;                         // b*16+g; group chans contiguous
    const float* p = x + (size_t)bg * (CPG * HW);
    float s = 0.f, ss = 0.f;
    for (int i = threadIdx.x; i < CPG * HW; i += 256) {
        float v = p[i]; s += v; ss += v * v;
    }
    for (int m = 1; m < 32; m <<= 1) {
        s  += __shfl_xor(s,  m, 32);
        ss += __shfl_xor(ss, m, 32);
    }
    __shared__ float as[8], ass[8];
    int w = threadIdx.x >> 5;
    if ((threadIdx.x & 31) == 0) { as[w] = s; ass[w] = ss; }
    __syncthreads();
    if (threadIdx.x == 0) {
        float S = 0.f, SS = 0.f;
        for (int i = 0; i < 8; i++) { S += as[i]; SS += ass[i]; }
        const float inv_n = 1.0f / (CPG * HW);
        float m_ = S * inv_n;
        float v_ = SS * inv_n - m_ * m_;
        mu[bg]   = m_;
        rstd[bg] = rsqrtf(v_ + 1e-5f);
    }
}

// ------------------------------------------------------- weight prep (f16)
__global__ __launch_bounds__(256) void prep_kernel(
        const float* __restrict__ qkv_w, const float* __restrict__ proj_w,
        _Float16* __restrict__ Wqkv, _Float16* __restrict__ Wpt) {
    int i = blockIdx.x * 256 + threadIdx.x;
    if (i < OC3 * CH) Wqkv[i] = (_Float16)qkv_w[i];
    if (i < CH * CH) {
        int o = i / CH, c = i % CH;
        Wpt[c * CH + o] = (_Float16)proj_w[i];    // transposed [c][o]
    }
}

// --------------------------------------- fused GN + QKV projection (WMMA)
// qkv[o][t] = sum_c W[o][c] * xn[c][t];  one 16x16 tile per wave
__global__ __launch_bounds__(256) void qkv_kernel(
        const float* __restrict__ x,
        const float* __restrict__ gn_w, const float* __restrict__ gn_b,
        const float* __restrict__ qkv_b,
        const float* __restrict__ mu, const float* __restrict__ rstd,
        const _Float16* __restrict__ Wqkv,
        _Float16* __restrict__ Qh, _Float16* __restrict__ Kth,
        _Float16* __restrict__ Vh) {
    int wave = threadIdx.x >> 5, lane = threadIdx.x & 31;
    int tile = blockIdx.x * 8 + wave;            // 4*48*256 tiles
    int b  = tile / (48 * 256);
    int r  = tile % (48 * 256);
    int ob = (r % 48) * 16;                      // output-channel tile base
    int tb = (r / 48) * 16;                      // token tile base
    int mrow = lane & 15, half = lane >> 4, kb = half * 8;

    const size_t xb = (size_t)b * CH * HW;
    v8f acc = {};
    #pragma unroll
    for (int ks = 0; ks < 8; ks++) {             // K = 256 in steps of 32
        int ck = ks * 32;
        // A fragment: W rows (lane = row o, K chunks kb / kb+16)
        H16 a;
        int o = ob + mrow;
        a.h2[0] = *(const v8h*)(Wqkv + (size_t)o * CH + ck + kb);
        a.h2[1] = *(const v8h*)(Wqkv + (size_t)o * CH + ck + kb + 16);
        // B fragment: lane = token col tb+mrow, K chunks kb / kb+16, GN fused
        const float* xr = x + xb + tb + mrow;
        H16 bb;
        #pragma unroll
        for (int j = 0; j < 16; j++) {
            int c = ck + kb + (j & 7) + ((j >> 3) << 4);
            int g = c >> 4;
            float m_ = mu[b * NGRP + g], r_ = rstd[b * NGRP + g];
            float scale = r_ * gn_w[c];
            float shift = gn_b[c] - m_ * scale;
            bb.e[j] = (_Float16)(xr[(size_t)c * HW] * scale + shift);
        }
        acc = wmma16(a.v, bb.v, acc);
    }
    // C layout: VGPR g -> row o = ob+g+8*half, col t = tb+(lane&15)
    int t = tb + mrow;
    #pragma unroll
    for (int g = 0; g < 8; g++) {
        int o = ob + g + 8 * half;
        float val = acc[g] + qkv_b[o];
        if (o < 256) {           // Q: token-major, fold 1/sqrt(c)=1/16
            Qh[((size_t)b * HW + t) * CH + o] = (_Float16)(val * 0.0625f);
        } else if (o < 512) {    // K: channel-major (ready as B operand)
            Kth[((size_t)b * CH + (o - 256)) * HW + t] = (_Float16)val;
        } else {                 // V: token-major (ready as B operand)
            Vh[((size_t)b * HW + t) * CH + (o - 512)] = (_Float16)val;
        }
    }
}

// ------------------------------------------------ flash attention (WMMA)
__global__ __launch_bounds__(256) void flash_kernel(
        const _Float16* __restrict__ Qh, const _Float16* __restrict__ Kth,
        const _Float16* __restrict__ Vh, float* __restrict__ O) {
    __shared__ __align__(32) _Float16 sK[256][40];   // chans x 32 keys (+pad)
    __shared__ __align__(32) _Float16 sV[32][264];   // keys  x 256 ch  (+pad)
    __shared__ __align__(32) _Float16 sP[QW][16][40];// per-wave P staging

    int wave = threadIdx.x >> 5, lane = threadIdx.x & 31;
    int b    = blockIdx.x / (HW / 16 / QW);          // 32 blocks per batch
    int qblk = blockIdx.x % (HW / 16 / QW);
    int qb   = (qblk * QW + wave) * 16;              // query tile base
    int mrow = lane & 15, half = lane >> 4, kb = half * 8;

    // Preload Q A-fragments (16 queries x 256 ch, already scaled by 1/16)
    const size_t qoff = ((size_t)b * HW + qb + mrow) * CH;
    H16 qf[8];
    #pragma unroll
    for (int ks = 0; ks < 8; ks++) {
        qf[ks].h2[0] = *(const v8h*)(Qh + qoff + ks * 32 + kb);
        qf[ks].h2[1] = *(const v8h*)(Qh + qoff + ks * 32 + kb + 16);
    }

    v8f zero = {};
    v8f accv[16];                                    // 16 ch-tiles x (16x16 f32)
    #pragma unroll
    for (int j = 0; j < 16; j++) accv[j] = zero;
    float rmax[8], rsum[8];
    #pragma unroll
    for (int g = 0; g < 8; g++) { rmax[g] = -3.0e38f; rsum[g] = 0.f; }

    const _Float16* Kb = Kth + (size_t)b * CH * HW;
    const _Float16* Vb = Vh  + (size_t)b * HW * CH;

    // per-thread staging coordinates (fixed across chunks)
    const int vr = threadIdx.x >> 3, vp = threadIdx.x & 7;
    const unsigned ldsK = (unsigned)(size_t)&sK[threadIdx.x][0];
    const unsigned ldsV = (unsigned)(size_t)&sV[vr][vp * 32];
    const _Float16* gK = Kb + (size_t)threadIdx.x * HW;     // row = channel
    const _Float16* gV = Vb + (size_t)vr * CH + vp * 32;    // row = key

    for (int kk = 0; kk < HW; kk += 32) {
        // ---- async DMA staging: K^T (256x32) and V (32x256) -> LDS
        async_b128(ldsK, (unsigned long long)(size_t)(gK + kk));
        async_b128(ldsV, (unsigned long long)(size_t)(gV + (size_t)kk * CH));
        if (kk + 32 < HW) {  // warm L2 for next chunk while we compute
            __builtin_prefetch(gK + kk + 32, 0, 3);
            __builtin_prefetch(gV + (size_t)(kk + 32) * CH, 0, 3);
        }
        asm volatile("s_wait_asynccnt 0" ::: "memory");
        __syncthreads();

        // ---- scores: S(16x32) = Q(16x256) @ K^T(256x32)
        v8f s0 = zero, s1 = zero;
        #pragma unroll
        for (int ks = 0; ks < 8; ks++) {
            int c = ks * 32 + kb;                    // B-frag: lane = key col, K chunks kb / kb+16
            H16 b0, b1;
            #pragma unroll
            for (int j = 0; j < 16; j++) {
                int cj = c + (j & 7) + ((j >> 3) << 4);
                b0.e[j] = sK[cj][mrow]; b1.e[j] = sK[cj][mrow + 16];
            }
            s0 = wmma16(qf[ks].v, b0.v, s0);
            s1 = wmma16(qf[ks].v, b1.v, s1);
        }

        // ---- online softmax (row stats per lane-half via 16-lane shuffles)
        float p0[8], p1[8], alpha[8];
        #pragma unroll
        for (int g = 0; g < 8; g++) {
            float t = fmaxf(s0[g], s1[g]);
            t = fmaxf(t, __shfl_xor(t, 1, 32));
            t = fmaxf(t, __shfl_xor(t, 2, 32));
            t = fmaxf(t, __shfl_xor(t, 4, 32));
            t = fmaxf(t, __shfl_xor(t, 8, 32));
            float mnew = fmaxf(rmax[g], t);
            alpha[g] = __expf(rmax[g] - mnew);
            rmax[g]  = mnew;
            p0[g] = __expf(s0[g] - mnew);
            p1[g] = __expf(s1[g] - mnew);
            float rs = p0[g] + p1[g];
            rs += __shfl_xor(rs, 1, 32);
            rs += __shfl_xor(rs, 2, 32);
            rs += __shfl_xor(rs, 4, 32);
            rs += __shfl_xor(rs, 8, 32);
            rsum[g] = rsum[g] * alpha[g] + rs;
        }
        #pragma unroll
        for (int j = 0; j < 16; j++)
            #pragma unroll
            for (int g = 0; g < 8; g++) accv[j][g] *= alpha[g];

        // ---- stage P (C-layout -> A-layout via per-wave LDS)
        #pragma unroll
        for (int g = 0; g < 8; g++) {
            sP[wave][g + 8 * half][mrow]      = (_Float16)p0[g];
            sP[wave][g + 8 * half][mrow + 16] = (_Float16)p1[g];
        }
        asm volatile("s_wait_dscnt 0" ::: "memory");   // LDS in-order per wave
        H16 pf;
        pf.h2[0] = *(const v8h*)&sP[wave][mrow][kb];
        pf.h2[1] = *(const v8h*)&sP[wave][mrow][kb + 16];

        // ---- accumulate: out += P(16x32) @ V(32x256)
        #pragma unroll
        for (int j = 0; j < 16; j++) {
            H16 vf;
            #pragma unroll
            for (int s = 0; s < 16; s++)
                vf.e[s] = sV[kb + (s & 7) + ((s >> 3) << 4)][j * 16 + mrow];
            accv[j] = wmma16(pf.v, vf.v, accv[j]);
        }
        __syncthreads();
    }

    // ---- epilogue: normalize and store O[b][t][c] (f32)
    float inv[8];
    #pragma unroll
    for (int g = 0; g < 8; g++) inv[g] = 1.0f / rsum[g];
    #pragma unroll
    for (int j = 0; j < 16; j++)
        #pragma unroll
        for (int g = 0; g < 8; g++) {
            int t = qb + g + 8 * half;
            O[((size_t)b * HW + t) * CH + j * 16 + mrow] = accv[j][g] * inv[g];
        }
}

// ------------------------------- proj matmul + bias + residual (WMMA)
// y[t][o] = sum_c O[t][c]*Wpt[c][o];  out[b][o][t] = x + y + pb
__global__ __launch_bounds__(256) void proj_kernel(
        const float* __restrict__ O, const _Float16* __restrict__ Wpt,
        const float* __restrict__ x, const float* __restrict__ proj_b,
        float* __restrict__ out) {
    int wave = threadIdx.x >> 5, lane = threadIdx.x & 31;
    int tile = blockIdx.x * 8 + wave;            // 4*16*256 tiles
    int b  = tile / (16 * 256);
    int r  = tile % (16 * 256);
    int ob = (r % 16) * 16;
    int tb = (r / 16) * 16;
    int mrow = lane & 15, half = lane >> 4, kb = half * 8;

    v8f acc = {};
    const float* Ob = O + ((size_t)b * HW + tb + mrow) * CH;   // A row = token
    #pragma unroll
    for (int ks = 0; ks < 8; ks++) {
        int ck = ks * 32;
        H16 a;
        #pragma unroll
        for (int j = 0; j < 8; j++) a.e[j]     = (_Float16)Ob[ck + kb + j];
        #pragma unroll
        for (int j = 0; j < 8; j++) a.e[8 + j] = (_Float16)Ob[ck + kb + 16 + j];
        int c = ck + kb;                         // B-frag: lane = col o, K chunks kb / kb+16
        H16 bb;
        #pragma unroll
        for (int j = 0; j < 16; j++)
            bb.e[j] = Wpt[(size_t)(c + (j & 7) + ((j >> 3) << 4)) * CH + ob + mrow];
        acc = wmma16(a.v, bb.v, acc);
    }
    // C: row t = tb+g+8*half (contiguous over g), col o = ob+(lane&15)
    int o = ob + mrow;
    float pb = proj_b[o];
    size_t base = ((size_t)b * CH + o) * HW + tb + 8 * half;
    v4f x0 = *(const v4f*)(x + base);
    v4f x1 = *(const v4f*)(x + base + 4);
    v4f r0, r1;
    #pragma unroll
    for (int g = 0; g < 4; g++) r0[g] = x0[g] + acc[g] + pb;
    #pragma unroll
    for (int g = 0; g < 4; g++) r1[g] = x1[g] + acc[4 + g] + pb;
    *(v4f*)(out + base)     = r0;
    *(v4f*)(out + base + 4) = r1;
}

// --------------------------------------------------------------- launcher
extern "C" void kernel_launch(void* const* d_in, const int* in_sizes, int n_in,
                              void* d_out, int out_size, void* d_ws, size_t ws_size,
                              hipStream_t stream) {
    const float* x      = (const float*)d_in[0];
    const float* gn_w   = (const float*)d_in[1];
    const float* gn_b   = (const float*)d_in[2];
    const float* qkv_w  = (const float*)d_in[3];
    const float* qkv_b  = (const float*)d_in[4];
    const float* proj_w = (const float*)d_in[5];
    const float* proj_b = (const float*)d_in[6];
    float* out = (float*)d_out;

    char* ws = (char*)d_ws;                      // ~41 MB total
    float*     mu   = (float*)(ws + 0);                       //   256 B
    float*     rstd = (float*)(ws + 256);                     //   256 B
    _Float16*  Wqkv = (_Float16*)(ws + 512);                  // 384 KB
    _Float16*  Wpt  = (_Float16*)(ws + 512 + 393216);         // 128 KB
    _Float16*  Qh   = (_Float16*)(ws + 524800);               // 8 MB
    _Float16*  Kth  = (_Float16*)(ws + 524800 + 8388608);     // 8 MB
    _Float16*  Vh   = (_Float16*)(ws + 524800 + 2*8388608ull);// 8 MB
    float*     O    = (float*)(ws + 524800 + 3*8388608ull);   // 16 MB

    gn_stats_kernel<<<64,   256, 0, stream>>>(x, mu, rstd);
    prep_kernel    <<<768,  256, 0, stream>>>(qkv_w, proj_w, Wqkv, Wpt);
    qkv_kernel     <<<6144, 256, 0, stream>>>(x, gn_w, gn_b, qkv_b, mu, rstd,
                                              Wqkv, Qh, Kth, Vh);
    flash_kernel   <<<128,  256, 0, stream>>>(Qh, Kth, Vh, O);
    proj_kernel    <<<2048, 256, 0, stream>>>(O, Wpt, x, proj_b, out);
}
